// VibeyMedicalAttention_19198503813293
// MI455X (gfx1250) — compile-verified, matrix operand order checked by reading, not run on hardware
//
#include <hip/hip_runtime.h>

// ---------------------------------------------------------------- types
typedef __attribute__((ext_vector_type(16))) __bf16        v16bf;
typedef __attribute__((ext_vector_type(8)))  float         v8f;
typedef __attribute__((ext_vector_type(4)))  unsigned int  v4u;

union FragBF {
  v16bf          v;
  v4u            q[2];
  unsigned short s[16];
};

__device__ __forceinline__ unsigned short f2bf(float f) {
  unsigned int u = __float_as_uint(f);
  u += 0x7FFFu + ((u >> 16) & 1u);   // round-to-nearest-even
  return (unsigned short)(u >> 16);
}

// A-fragment (and B-fragment, since B is stored [N,K]) loader for
// v_wmma_f32_16x16x32_bf16. Per CDNA5 ISA 16-bit A layout:
// lane l holds row (l&15); lanes 0-15 carry K = k0+[0..7] and k0+16+[0..7],
// lanes 16-31 carry K = k0+8+[0..7] and k0+24+[0..7]; two b128 loads.
__device__ __forceinline__ v16bf load_frag_bf16(const unsigned short* base,
                                                int ld, int rc, int k0, int lane) {
  int r  = rc + (lane & 15);
  int kh = (lane >> 4) * 8;
  const unsigned short* p = base + (long)r * ld + k0;
  FragBF f;
  f.q[0] = *(const v4u*)(p + kh);
  f.q[1] = *(const v4u*)(p + 16 + kh);
  return f.v;
}

// Same layout but source is f32 (probs); convert to bf16 in registers.
__device__ __forceinline__ v16bf load_frag_f32(const float* base,
                                               int ld, int rc, int k0, int lane) {
  int r  = rc + (lane & 15);
  int kh = (lane >> 4) * 8;
  const float* p = base + (long)r * ld + k0;
  FragBF f;
#pragma unroll
  for (int j = 0; j < 8; ++j) f.s[j]     = f2bf(p[kh + j]);
#pragma unroll
  for (int j = 0; j < 8; ++j) f.s[8 + j] = f2bf(p[16 + kh + j]);
  return f.v;
}

__device__ __forceinline__ v8f wmma_bf16(v16bf a, v16bf b, v8f c) {
  return __builtin_amdgcn_wmma_f32_16x16x32_bf16(false, a, false, b,
                                                 (short)0, c, false, false);
}

// ---------------------------------------------------------------- constants
#define CB 2
#define CS 2048
#define CH 1024
#define CNH 16
#define CD 64
#define CM 512
#define CHM 1536
#define CBS 4096   // B*S

// ---------------------------------------------------------------- conversions
__global__ void vma_cvt_kernel(const float* __restrict__ src,
                               unsigned short* __restrict__ dst, long n) {
  long i = (long)blockIdx.x * 256 + threadIdx.x;
  if (i < n) dst[i] = f2bf(src[i]);
}

__global__ void vma_cvtT_kernel(const float* __restrict__ src,
                                unsigned short* __restrict__ dst, int R, int C) {
  long i = (long)blockIdx.x * 256 + threadIdx.x;
  if (i < (long)R * C) {
    int r = (int)(i / C), c = (int)(i % C);
    dst[(long)c * R + r] = f2bf(src[i]);
  }
}

// v [B*S, H] bf16 -> vT [B, NH, D, S] bf16
__global__ void vma_vtrans_kernel(const unsigned short* __restrict__ v,
                                  unsigned short* __restrict__ vT) {
  long i = (long)blockIdx.x * 256 + threadIdx.x;
  if (i >= (long)CBS * CH) return;
  long row = i / CH; int c = (int)(i % CH);
  int b = (int)(row / CS), s = (int)(row % CS);
  int h = c / CD, d = c % CD;
  vT[(((long)(b * CNH + h) * CD + d) * CS) + s] = v[i];
}

// ---------------------------------------------------------------- generic GEMM
// C[m,n] = sum_k A[m,k]*Bt[n,k] (+bias[n]); optional f32 and/or bf16 outputs.
// 256 threads = 8 waves (4x2), 4x2 WMMA tiles per wave -> 256x64 WG tile.
// 8 wmma per 12 b128 loads per k-step.
__global__ __launch_bounds__(256)
void vma_gemm_kernel(const unsigned short* __restrict__ A, int lda, long aBatch,
                     const unsigned short* __restrict__ Bt, int ldb, long bBatch,
                     const float* __restrict__ bias,
                     float* __restrict__ Cf, unsigned short* __restrict__ Cb,
                     int ldc, long cBatch, int K) {
  int lane = threadIdx.x & 31;
  int wave = threadIdx.x >> 5;
  int wm = wave >> 1, wn = wave & 1;
  long zb = blockIdx.z;
  const unsigned short* a = A + zb * aBatch;
  const unsigned short* b = Bt + zb * bBatch;
  int row0 = blockIdx.x * 256 + wm * 64;
  int col0 = blockIdx.y * 64 + wn * 32;
  v8f acc[4][2] = {};
  for (int k0 = 0; k0 < K; k0 += 32) {
    v16bf af[4];
#pragma unroll
    for (int i = 0; i < 4; ++i) af[i] = load_frag_bf16(a, lda, row0 + i * 16, k0, lane);
    v16bf b0 = load_frag_bf16(b, ldb, col0,      k0, lane);
    v16bf b1 = load_frag_bf16(b, ldb, col0 + 16, k0, lane);
#pragma unroll
    for (int i = 0; i < 4; ++i) {
      acc[i][0] = wmma_bf16(af[i], b0, acc[i][0]);
      acc[i][1] = wmma_bf16(af[i], b1, acc[i][1]);
    }
  }
  int ln15 = lane & 15;
  int mo   = (lane >> 4) * 8;
#pragma unroll
  for (int i = 0; i < 4; ++i)
#pragma unroll
    for (int j = 0; j < 2; ++j) {
      int n = col0 + j * 16 + ln15;
      float bv = bias ? bias[n] : 0.0f;
#pragma unroll
      for (int e = 0; e < 8; ++e) {
        int m = row0 + i * 16 + mo + e;
        float val = acc[i][j][e] + bv;
        long idx = zb * cBatch + (long)m * ldc + n;
        if (Cf) Cf[idx] = val;
        if (Cb) Cb[idx] = f2bf(val);
      }
    }
}

// ---------------------------------------------------------------- fused scores
// One WG owns a full 16-row x 2048-col strip of scores for one (b,h):
//   scores = (q.kT)/8 + 0.3*med_softmax, then full row softmax, written ONCE.
// Wave w covers cols [w*256, w*256+256) with 16 accumulators (K=64 -> 2 steps).
// Row r = e + 8*(lane>=16) lives entirely in one 16-lane half, so the row
// reduction is 4 shfl_xor steps + an 8-wave LDS combine.
__global__ __launch_bounds__(256)
void vma_score_softmax_kernel(const unsigned short* __restrict__ qb,
                              const unsigned short* __restrict__ kb,
                              const float* __restrict__ medp,
                              float* __restrict__ probs) {
  int lane = threadIdx.x & 31;
  int wave = threadIdx.x >> 5;
  int bh = blockIdx.z;
  int bz = bh / CNH, h = bh % CNH;
  const unsigned short* aq = qb + (long)bz * CS * CH + h * CD;
  const unsigned short* bk = kb + (long)bz * CS * CH + h * CD;
  int row0    = blockIdx.x * 16;
  int colbase = wave * 256;

  v8f acc[16] = {};
#pragma unroll
  for (int k0 = 0; k0 < CD; k0 += 32) {
    v16bf a = load_frag_bf16(aq, CH, row0, k0, lane);
#pragma unroll
    for (int j = 0; j < 16; ++j) {
      v16bf bf = load_frag_bf16(bk, CH, colbase + j * 16, k0, lane);
      acc[j] = wmma_bf16(a, bf, acc[j]);
    }
  }

  int ln15 = lane & 15;
  int mo   = (lane >> 4) * 8;
  const float* medb = medp + (long)bz * CS * CS;
  float* out = probs + (long)bh * CS * CS;

  // scale + med bias, per-row running max
  float mr[8];
#pragma unroll
  for (int e = 0; e < 8; ++e) mr[e] = -3.4e38f;
#pragma unroll
  for (int j = 0; j < 16; ++j) {
    int n = colbase + j * 16 + ln15;
#pragma unroll
    for (int e = 0; e < 8; ++e) {
      int m = row0 + mo + e;
      float v = acc[j][e] * 0.125f + 0.3f * medb[(long)m * CS + n];
      acc[j][e] = v;
      mr[e] = fmaxf(mr[e], v);
    }
  }
#pragma unroll
  for (int e = 0; e < 8; ++e)
    for (int msk = 1; msk < 16; msk <<= 1)
      mr[e] = fmaxf(mr[e], __shfl_xor(mr[e], msk, 32));

  __shared__ float smax[8][16];
  __shared__ float ssum[8][16];
  if (ln15 == 0) {
#pragma unroll
    for (int e = 0; e < 8; ++e) smax[wave][mo + e] = mr[e];
  }
  __syncthreads();
  float rmax[8];
#pragma unroll
  for (int e = 0; e < 8; ++e) {
    float v = smax[0][mo + e];
#pragma unroll
    for (int w = 1; w < 8; ++w) v = fmaxf(v, smax[w][mo + e]);
    rmax[e] = v;
  }

  // exp + per-row running sum
  float sr[8];
#pragma unroll
  for (int e = 0; e < 8; ++e) sr[e] = 0.0f;
#pragma unroll
  for (int j = 0; j < 16; ++j)
#pragma unroll
    for (int e = 0; e < 8; ++e) {
      float v = __expf(acc[j][e] - rmax[e]);
      acc[j][e] = v;
      sr[e] += v;
    }
#pragma unroll
  for (int e = 0; e < 8; ++e)
    for (int msk = 1; msk < 16; msk <<= 1)
      sr[e] += __shfl_xor(sr[e], msk, 32);
  if (ln15 == 0) {
#pragma unroll
    for (int e = 0; e < 8; ++e) ssum[wave][mo + e] = sr[e];
  }
  __syncthreads();
#pragma unroll
  for (int e = 0; e < 8; ++e) {
    float v = ssum[0][mo + e];
#pragma unroll
    for (int w = 1; w < 8; ++w) v += ssum[w][mo + e];
    sr[e] = 1.0f / v;
  }

  // single write of normalized probabilities
#pragma unroll
  for (int j = 0; j < 16; ++j) {
    int n = colbase + j * 16 + ln15;
#pragma unroll
    for (int e = 0; e < 8; ++e) {
      int m = row0 + mo + e;
      out[(long)m * CS + n] = acc[j][e] * sr[e];
    }
  }
}

// ---------------------------------------------------------------- P.V kernel
// ctx[b,s,h*64+d] = sum_t probs[b,h,s,t]*v[b,t,h*64+d]; probs cvt'd in regs.
__global__ __launch_bounds__(256)
void vma_av_kernel(const float* __restrict__ probs,
                   const unsigned short* __restrict__ vT,
                   unsigned short* __restrict__ cat) {
  int lane = threadIdx.x & 31;
  int wave = threadIdx.x >> 5;
  int wm = wave >> 1, wn = wave & 1;
  int bh = blockIdx.z;
  int bz = bh / CNH, h = bh % CNH;
  const float*          a = probs + (long)bh * CS * CS;   // [S,S], lda=S
  const unsigned short* b = vT    + (long)bh * CD * CS;   // [D,S], ldb=S
  int row0 = blockIdx.x * 128 + wm * 32;
  int col0 = wn * 32;                                     // N = D = 64
  v8f acc[2][2] = {};
  for (int k0 = 0; k0 < CS; k0 += 32) {
    v16bf a0 = load_frag_f32(a, CS, row0,      k0, lane);
    v16bf a1 = load_frag_f32(a, CS, row0 + 16, k0, lane);
    v16bf b0 = load_frag_bf16(b, CS, col0,      k0, lane);
    v16bf b1 = load_frag_bf16(b, CS, col0 + 16, k0, lane);
    acc[0][0] = wmma_bf16(a0, b0, acc[0][0]);
    acc[0][1] = wmma_bf16(a0, b1, acc[0][1]);
    acc[1][0] = wmma_bf16(a1, b0, acc[1][0]);
    acc[1][1] = wmma_bf16(a1, b1, acc[1][1]);
  }
  int ln15 = lane & 15;
  int mo   = (lane >> 4) * 8;
#pragma unroll
  for (int i = 0; i < 2; ++i)
#pragma unroll
    for (int j = 0; j < 2; ++j) {
      int n = col0 + j * 16 + ln15;
#pragma unroll
      for (int e = 0; e < 8; ++e) {
        int m = row0 + i * 16 + mo + e;
        long row = (long)bz * CS + m;
        cat[row * CHM + h * CD + n] = f2bf(acc[i][j][e]);
      }
    }
}

// ------------------------------------------------ register-resident softmax
// cols fixed at 2048: 8 floats/thread in registers -> exactly 1R + 1W.
__global__ __launch_bounds__(256)
void vma_softmax2048_kernel(float* __restrict__ data) {
  long row = blockIdx.x;
  float* p = data + row * (long)CS;
  __shared__ float red[256];
  int t = threadIdx.x;
  float x[8], mx = -3.4e38f;
#pragma unroll
  for (int j = 0; j < 8; ++j) { x[j] = p[t + 256 * j]; mx = fmaxf(mx, x[j]); }
  red[t] = mx; __syncthreads();
  for (int o = 128; o; o >>= 1) { if (t < o) red[t] = fmaxf(red[t], red[t + o]); __syncthreads(); }
  mx = red[0]; __syncthreads();
  float s = 0.0f;
#pragma unroll
  for (int j = 0; j < 8; ++j) { x[j] = __expf(x[j] - mx); s += x[j]; }
  red[t] = s; __syncthreads();
  for (int o = 128; o; o >>= 1) { if (t < o) red[t] += red[t + o]; __syncthreads(); }
  float inv = 1.0f / red[0];
#pragma unroll
  for (int j = 0; j < 8; ++j) p[t + 256 * j] = x[j] * inv;
}

// ---------------------------------------------------------------- residual + LN
__global__ __launch_bounds__(256)
void vma_ln_kernel(const float* __restrict__ pre, const float* __restrict__ hs,
                   const float* __restrict__ g, const float* __restrict__ bb,
                   float* __restrict__ out) {
  long row = blockIdx.x;
  const float* pp = pre + row * CH;
  const float* hh = hs  + row * CH;
  __shared__ float red[256];
  int t = threadIdx.x;
  float x[4];
  float sum = 0.0f;
#pragma unroll
  for (int j = 0; j < 4; ++j) { x[j] = pp[t + j * 256] + hh[t + j * 256]; sum += x[j]; }
  red[t] = sum; __syncthreads();
  for (int o = 128; o; o >>= 1) { if (t < o) red[t] += red[t + o]; __syncthreads(); }
  float mu = red[0] * (1.0f / CH); __syncthreads();
  float vs = 0.0f;
#pragma unroll
  for (int j = 0; j < 4; ++j) { float d = x[j] - mu; vs += d * d; }
  red[t] = vs; __syncthreads();
  for (int o = 128; o; o >>= 1) { if (t < o) red[t] += red[t + o]; __syncthreads(); }
  float inv = rsqrtf(red[0] * (1.0f / CH) + 1e-12f);
#pragma unroll
  for (int j = 0; j < 4; ++j) {
    int c = t + j * 256;
    out[row * CH + c] = (x[j] - mu) * inv * g[c] + bb[c];
  }
}

// ---------------------------------------------------------------- host launch
extern "C" void kernel_launch(void* const* d_in, const int* in_sizes, int n_in,
                              void* d_out, int out_size, void* d_ws, size_t ws_size,
                              hipStream_t stream) {
  (void)in_sizes; (void)n_in; (void)out_size; (void)ws_size;
  const float* hs   = (const float*)d_in[0];
  const float* mctx = (const float*)d_in[1];
  const float* Wq   = (const float*)d_in[2];
  const float* bq   = (const float*)d_in[3];
  const float* Wk   = (const float*)d_in[4];
  const float* bk   = (const float*)d_in[5];
  const float* Wv   = (const float*)d_in[6];
  const float* bv   = (const float*)d_in[7];
  const float* Wmq  = (const float*)d_in[8];
  const float* bmq  = (const float*)d_in[9];
  const float* Wmk  = (const float*)d_in[10];
  const float* bmk  = (const float*)d_in[11];
  const float* Wc   = (const float*)d_in[12];
  const float* bc   = (const float*)d_in[13];
  const float* Wf   = (const float*)d_in[14];
  const float* bf   = (const float*)d_in[15];
  const float* lng  = (const float*)d_in[16];
  const float* lnb  = (const float*)d_in[17];

  float* out   = (float*)d_out;
  float* probs = out + (size_t)CBS * CH;   // [B,NH,S,S] f32, second output

  char* ws = (char*)d_ws;
  size_t off = 0;
  auto alloc = [&](size_t bytes) -> void* {
    void* p = ws + off;
    off += (bytes + 255) & ~(size_t)255;
    return p;
  };
  typedef unsigned short u16;
  u16* hsb   = (u16*)alloc((size_t)CBS * CH * 2);
  u16* medb  = (u16*)alloc((size_t)CBS * CH * 2);
  u16* WqT   = (u16*)alloc((size_t)CH * CH * 2);
  u16* WkT   = (u16*)alloc((size_t)CH * CH * 2);
  u16* WvT   = (u16*)alloc((size_t)CH * CH * 2);
  u16* WmqT  = (u16*)alloc((size_t)CM * CH * 2);
  u16* WmkT  = (u16*)alloc((size_t)CM * CH * 2);
  u16* WcT   = (u16*)alloc((size_t)CM * CH * 2);
  u16* WfT   = (u16*)alloc((size_t)CH * CHM * 2);
  u16* qb    = (u16*)alloc((size_t)CBS * CH * 2);
  u16* kb    = (u16*)alloc((size_t)CBS * CH * 2);
  u16* vb    = (u16*)alloc((size_t)CBS * CH * 2);
  u16* vT    = (u16*)alloc((size_t)CBS * CH * 2);
  u16* mqb   = (u16*)alloc((size_t)CBS * CM * 2);
  u16* mkb   = (u16*)alloc((size_t)CBS * CM * 2);
  float* medp = (float*)alloc((size_t)CB * CS * CS * 4);
  u16* cat   = (u16*)alloc((size_t)CBS * CHM * 2);
  float* pre = (float*)alloc((size_t)CBS * CH * 4);

  dim3 blk(256);
  long nHS = (long)CBS * CH;
  // f32 -> bf16 activations
  vma_cvt_kernel<<<dim3((unsigned)((nHS + 255) / 256)), blk, 0, stream>>>(hs, hsb, nHS);
  vma_cvt_kernel<<<dim3((unsigned)((nHS + 255) / 256)), blk, 0, stream>>>(mctx, medb, nHS);
  // weights -> bf16 transposed [N,K]
  vma_cvtT_kernel<<<dim3((CH * CH + 255) / 256), blk, 0, stream>>>(Wq, WqT, CH, CH);
  vma_cvtT_kernel<<<dim3((CH * CH + 255) / 256), blk, 0, stream>>>(Wk, WkT, CH, CH);
  vma_cvtT_kernel<<<dim3((CH * CH + 255) / 256), blk, 0, stream>>>(Wv, WvT, CH, CH);
  vma_cvtT_kernel<<<dim3((CH * CM + 255) / 256), blk, 0, stream>>>(Wmq, WmqT, CH, CM);
  vma_cvtT_kernel<<<dim3((CH * CM + 255) / 256), blk, 0, stream>>>(Wmk, WmkT, CH, CM);
  vma_cvtT_kernel<<<dim3((CH * CM + 255) / 256), blk, 0, stream>>>(Wc, WcT, CH, CM);
  vma_cvtT_kernel<<<dim3((CHM * CH + 255) / 256), blk, 0, stream>>>(Wf, WfT, CHM, CH);

  // QKV projections (bf16 out)
  dim3 gQ(CBS / 256, CH / 64, 1);
  vma_gemm_kernel<<<gQ, blk, 0, stream>>>(hsb, CH, 0, WqT, CH, 0, bq, nullptr, qb, CH, 0, CH);
  vma_gemm_kernel<<<gQ, blk, 0, stream>>>(hsb, CH, 0, WkT, CH, 0, bk, nullptr, kb, CH, 0, CH);
  vma_gemm_kernel<<<gQ, blk, 0, stream>>>(hsb, CH, 0, WvT, CH, 0, bv, nullptr, vb, CH, 0, CH);
  // medical projections
  dim3 gM(CBS / 256, CM / 64, 1);
  vma_gemm_kernel<<<gM, blk, 0, stream>>>(hsb,  CH, 0, WmqT, CH, 0, bmq, nullptr, mqb, CM, 0, CH);
  vma_gemm_kernel<<<gM, blk, 0, stream>>>(medb, CH, 0, WmkT, CH, 0, bmk, nullptr, mkb, CM, 0, CH);
  // med logits (batched over B), then 1R+1W register softmax
  dim3 gMed(CS / 256, CS / 64, CB);
  vma_gemm_kernel<<<gMed, blk, 0, stream>>>(mqb, CM, (long)CS * CM, mkb, CM, (long)CS * CM,
                                            nullptr, medp, nullptr, CS, (long)CS * CS, CM);
  vma_softmax2048_kernel<<<dim3(CB * CS), blk, 0, stream>>>(medp);
  // v -> [B,NH,D,S]
  vma_vtrans_kernel<<<dim3((unsigned)((nHS + 255) / 256)), blk, 0, stream>>>(vb, vT);
  // fused q.kT/8 + 0.3*med + row softmax -> probs written exactly once
  dim3 gS(CS / 16, 1, CB * CNH);
  vma_score_softmax_kernel<<<gS, blk, 0, stream>>>(qb, kb, medp, probs);
  // ctx = probs @ v  (f32 probs cvt'd in registers) -> cat[:, 0:1024]
  dim3 gAV(CS / 128, 1, CB * CNH);
  vma_av_kernel<<<gAV, blk, 0, stream>>>(probs, vT, cat);
  // clin = ctx @ Wc + bc -> cat[:, 1024:1536]
  dim3 gC(CBS / 256, CM / 64, 1);
  vma_gemm_kernel<<<gC, blk, 0, stream>>>(cat, CHM, 0, WcT, CH, 0, bc,
                                          nullptr, cat + CH, CHM, 0, CH);
  // pre = [ctx|clin] @ Wf + bf  (single K=1536 GEMM)
  dim3 gF(CBS / 256, CH / 64, 1);
  vma_gemm_kernel<<<gF, blk, 0, stream>>>(cat, CHM, 0, WfT, CHM, 0, bf,
                                          pre, nullptr, CH, 0, CHM);
  // residual + layernorm -> out
  vma_ln_kernel<<<dim3(CBS), blk, 0, stream>>>(pre, hs, lng, lnb, out);
}
